// ConvectionModule_76081050681508
// MI455X (gfx1250) — matrix operand order repaired and checked by reading, not compiled, not run on hardware
//
#include <hip/hip_runtime.h>
#include <hip/hip_bf16.h>

typedef __attribute__((ext_vector_type(16))) _Float16 v16h;
typedef __attribute__((ext_vector_type(2)))  _Float16 v2h;
typedef __attribute__((ext_vector_type(2)))  __fp16   v2fp;
typedef __attribute__((ext_vector_type(8)))  float    v8f;

#define NN 8192
#define DD 512

union H16 { v16h h; v2h h2[8]; float4 q[2]; };

// v_cvt_pk_rtz_f16_f32 with the vector element type clang expects
__device__ __forceinline__ v2h pkrtz(float a, float b) {
  union { v2fp f; v2h h; } u;
  u.f = __builtin_amdgcn_cvt_pkrtz(a, b);
  return u.h;
}

// 16 contiguous halfs (unused; B operands are loaded with ld_split)
__device__ __forceinline__ v16h ld_cont(const _Float16* p) {
  H16 u;
  u.q[0] = *(const float4*)(p);
  u.q[1] = *(const float4*)(p + 8);
  return u.h;
}
// A operand: halfs [0..7] at p, halfs [8..15] at p+16 (K = b..b+7, b+16..b+23)
__device__ __forceinline__ v16h ld_split(const _Float16* p) {
  H16 u;
  u.q[0] = *(const float4*)(p);
  u.q[1] = *(const float4*)(p + 16);
  return u.h;
}

// ---- Kernel 1: s = Z @ W_V ; E = exp(s) ; Zh = f16(Z). One wave per row. ----
__global__ void conv_prep(const float* __restrict__ Z, const float* __restrict__ WV,
                          _Float16* __restrict__ Zh, float* __restrict__ E) {
  int wid  = threadIdx.x >> 5;
  int lane = threadIdx.x & 31;
  int i = blockIdx.x * 8 + wid;
  const float* zr = Z + (size_t)i * DD;
  v2h*         zo = (v2h*)(Zh + (size_t)i * DD);
  float part = 0.f;
#pragma unroll
  for (int t = 0; t < DD / 64; ++t) {
    int k = 2 * lane + 64 * t;
    float z0 = zr[k], z1 = zr[k + 1];
    part = fmaf(z0, WV[k], part);
    part = fmaf(z1, WV[k + 1], part);
    zo[k >> 1] = pkrtz(z0, z1);
  }
#pragma unroll
  for (int off = 16; off > 0; off >>= 1)
    part += __shfl_xor(part, off, 32);
  if (lane == 0) E[i] = __expf(part);
}

// ---- Kernel 2: W_C -> f16 (packed pairs) ----
__global__ void conv_cvt_wc(const float* __restrict__ W, _Float16* __restrict__ Wh) {
  int t = blockIdx.x * blockDim.x + threadIdx.x;
  ((v2h*)Wh)[t] = pkrtz(W[2 * t], W[2 * t + 1]);
}

// ---- Kernel 3: Vt[d][i] = sum_k Zh[i][k] * WCh[d][k]  (WMMA GEMM, Vt = V^T f16) ----
// wave -> 16 d-rows x 64 i-cols (4 tiles). 4096 waves. Double-buffered k-loop.
__global__ void conv_gemmV(const _Float16* __restrict__ Zh, const _Float16* __restrict__ WCh,
                           _Float16* __restrict__ Vt) {
  int wid  = threadIdx.x >> 5;
  int lane = threadIdx.x & 31;
  int W  = blockIdx.x * 8 + wid;          // 0..4095
  int d0 = (W >> 7) * 16;                 // 32 d-tiles
  int i0 = (W & 127) * 64;                // 128 i-blocks
  int half = lane >> 4, m = lane & 15;
  int abase = half * 8;                   // A K-base
  int bbase = half * 8;                   // B K-base
  v8f acc[4] = {};
  const _Float16* ap = WCh + (size_t)(d0 + m) * DD + abase;
  const _Float16* bp = Zh  + (size_t)(i0 + m) * DD + bbase;

  v16h a0, a1, b0[4], b1[4];
  a0 = ld_split(ap);
#pragma unroll
  for (int t = 0; t < 4; ++t) b0[t] = ld_split(bp + (size_t)t * 16 * DD);

#pragma unroll 1
  for (int k0 = 0; k0 < DD; k0 += 64) {
    int k1 = k0 + 32;
    a1 = ld_split(ap + k1);
#pragma unroll
    for (int t = 0; t < 4; ++t) b1[t] = ld_split(bp + (size_t)t * 16 * DD + k1);
#pragma unroll
    for (int t = 0; t < 4; ++t)
      acc[t] = __builtin_amdgcn_wmma_f32_16x16x32_f16(false, a0, false, b0[t], (short)0, acc[t], false, false);

    int k2 = (k0 + 64) & (DD - 1);        // wraps to 0 on last iter (dummy prefetch)
    a0 = ld_split(ap + k2);
#pragma unroll
    for (int t = 0; t < 4; ++t) b0[t] = ld_split(bp + (size_t)t * 16 * DD + k2);
#pragma unroll
    for (int t = 0; t < 4; ++t)
      acc[t] = __builtin_amdgcn_wmma_f32_16x16x32_f16(false, a1, false, b1[t], (short)0, acc[t], false, false);
  }

  _Float16* op = Vt + (size_t)(d0 + half * 8) * NN + i0 + m;
#pragma unroll
  for (int r = 0; r < 8; ++r) {
    op[(size_t)r * NN +  0] = (_Float16)acc[0][r];
    op[(size_t)r * NN + 16] = (_Float16)acc[1][r];
    op[(size_t)r * NN + 32] = (_Float16)acc[2][r];
    op[(size_t)r * NN + 48] = (_Float16)acc[3][r];
  }
}

// ---- Kernel 4 helpers ----
__device__ __forceinline__ void loadB16(const _Float16* vb, int j0, v16h (&b)[16]) {
#pragma unroll
  for (int t = 0; t < 16; ++t)
    b[t] = ld_split(vb + (size_t)t * 16 * NN + j0);
}
__device__ __forceinline__ void loadE4(const float* E, int j0, int abase, float4 (&e)[4]) {
  const float4* p = (const float4*)(E + j0 + abase);
  e[0] = p[0]; e[1] = p[1];
  const float4* q = (const float4*)(E + j0 + abase + 16);
  e[2] = q[0]; e[3] = q[1];
}
// Build the w A-tile: w = exp(sigmoid(s_i - s_j)) = exp(1/(1 + E_j * E_i^-1))
// Packed pair conversion (v_cvt_pk_rtz_f16_f32) to minimize VALU under the WMMAs.
__device__ __forceinline__ v16h mkA(const float4 (&e)[4], float invEi, float& p) {
  float ej[16] = { e[0].x, e[0].y, e[0].z, e[0].w, e[1].x, e[1].y, e[1].z, e[1].w,
                   e[2].x, e[2].y, e[2].z, e[2].w, e[3].x, e[3].y, e[3].z, e[3].w };
  H16 A;
#pragma unroll
  for (int i = 0; i < 8; ++i) {
    float u0 = ej[2 * i]     * invEi;
    float u1 = ej[2 * i + 1] * invEi;
    float s0 = __builtin_amdgcn_rcpf(1.0f + u0);
    float s1 = __builtin_amdgcn_rcpf(1.0f + u1);
    float w0 = __expf(s0);
    float w1 = __expf(s1);
    p += w0 + w1;
    A.h2[i] = pkrtz(w0, w1);
  }
  return A.h;
}

// ---- Kernel 4: fused flash-style  out_i = (sum_j w_ij V_j) / (sum_j w_ij) ----
// wave -> 16 i-rows x 256 d-cols (16 tiles). 1024 independent waves, no barriers.
// Register double-buffered over j so loads for step k+1 fly under WMMAs of step k.
__global__ void conv_flash(const float* __restrict__ E, const _Float16* __restrict__ Vt,
                           float* __restrict__ out) {
  int wid  = threadIdx.x >> 5;
  int lane = threadIdx.x & 31;
  int W  = blockIdx.x * 8 + wid;          // 0..1023
  int i0 = (W >> 1) * 16;
  int d0 = (W & 1) * 256;
  int half = lane >> 4, m = lane & 15;
  int abase = half * 8;
  float invEi = 1.0f / E[i0 + m];
  v8f acc[16] = {};
  float p = 0.f;
  const _Float16* vb = Vt + (size_t)(d0 + m) * NN + half * 8;

  v16h b0[16], b1[16];
  float4 e0[4], e1[4];
  loadE4(E, 0, abase, e0);
  loadB16(vb, 0, b0);

#pragma unroll 1
  for (int j0 = 0; j0 < NN; j0 += 64) {
    int j1 = j0 + 32;
    loadE4(E, j1, abase, e1);             // prefetch stage 1
    loadB16(vb, j1, b1);
    v16h a = mkA(e0, invEi, p);           // consumes loads from previous step
#pragma unroll
    for (int t = 0; t < 16; ++t)
      acc[t] = __builtin_amdgcn_wmma_f32_16x16x32_f16(false, a, false, b0[t], (short)0, acc[t], false, false);

    int j2 = (j0 + 64) & (NN - 1);        // wraps on last iter (dummy prefetch)
    loadE4(E, j2, abase, e0);             // prefetch stage 0
    loadB16(vb, j2, b0);
    v16h a2 = mkA(e1, invEi, p);
#pragma unroll
    for (int t = 0; t < 16; ++t)
      acc[t] = __builtin_amdgcn_wmma_f32_16x16x32_f16(false, a2, false, b1[t], (short)0, acc[t], false, false);
  }

  // row denominators: lane L covers K-halves {b..b+7,b+16..b+23}; pair L^16 covers the rest
  float pc = p + __shfl_xor(p, 16, 32);   // lanes m and m+16 now hold rowsum[m]
  float* op = out + (size_t)(i0 + half * 8) * DD + d0 + m;
#pragma unroll
  for (int r = 0; r < 8; ++r) {
    float rs  = __shfl(pc, r + 8 * half, 32);  // rowsum for C-row M = r + 8*half
    float inv = 1.0f / rs;
#pragma unroll
    for (int t = 0; t < 16; ++t)
      op[(size_t)r * DD + 16 * t] = acc[t][r] * inv;
  }
}

extern "C" void kernel_launch(void* const* d_in, const int* in_sizes, int n_in,
                              void* d_out, int out_size, void* d_ws, size_t ws_size,
                              hipStream_t stream) {
  const float* Z  = (const float*)d_in[0];   // [8192,512]
  const float* WC = (const float*)d_in[1];   // [512,512]
  const float* WV = (const float*)d_in[2];   // [512]
  float* out = (float*)d_out;                // [8192,512]

  char* base = (char*)d_ws;
  _Float16* Vt  = (_Float16*)(base);                          // 8 MB  [512][8192]
  _Float16* Zh  = (_Float16*)(base + (size_t)NN * DD * 2);    // 8 MB  [8192][512]
  _Float16* WCh = (_Float16*)(base + (size_t)NN * DD * 4);    // 512 KB [512][512]
  float*    E   = (float*)   (base + (size_t)NN * DD * 4 + (size_t)DD * DD * 2); // 32 KB

  conv_prep  <<<NN / 8, 256, 0, stream>>>(Z, WV, Zh, E);
  conv_cvt_wc<<<(DD * DD) / 512, 256, 0, stream>>>(WC, WCh);
  conv_gemmV <<<512, 256, 0, stream>>>(Zh, WCh, Vt);
  conv_flash <<<128, 256, 0, stream>>>(E, Vt, out);
}
